// TemporalAttention_24713241821724
// MI455X (gfx1250) — matrix operand order repaired and checked by reading, not compiled, not run on hardware
//
#include <hip/hip_runtime.h>

// ---------------------------------------------------------------------------
// TemporalAttention on MI455X (gfx1250, wave32, WMMA).
// Pipeline: [add emb + cvt f16] -> [cvt weights f16] -> [QKV WMMA GEMM with
// head reorg + Q prescale + V transpose] -> [flash attention, WMMA + online
// softmax] -> [proj WMMA GEMM, f32 out].
// GEMM kernels use 2x2 register macro-tiles (32x32 out per wave): 8 b128
// loads per 4 WMMAs instead of 4 loads per 1 WMMA.
// ---------------------------------------------------------------------------

typedef _Float16 v16h __attribute__((ext_vector_type(16)));
typedef _Float16 v8h  __attribute__((ext_vector_type(8)));
typedef float    v8f  __attribute__((ext_vector_type(8)));

#define C_DIM   1024
#define THREEC  3072
#define N_TOK   3072   // T*H*W = 12*256
#define B_DIM   2
#define NH      16
#define HD      64
#define M_TOT   6144   // B * N_TOK

// Build a 16-element f16 fragment from two 16-byte chunks.
static __device__ __forceinline__ v16h make_frag(const _Float16* p0,
                                                 const _Float16* p1) {
  v8h lo = *(const v8h*)p0;
  v8h hi = *(const v8h*)p1;
  v16h r;
#pragma unroll
  for (int i = 0; i < 8; ++i) { r[i] = lo[i]; r[i + 8] = hi[i]; }
  return r;
}
// B fragment in table order: elements 0..7 from p, 8..15 from p + 16.
static __device__ __forceinline__ v16h make_frag32(const _Float16* p) {
  return make_frag(p, p + 16);
}

static __device__ __forceinline__ v8f wmma16(v16h a, v16h b, v8f c) {
  return __builtin_amdgcn_wmma_f32_16x16x32_f16(
      /*neg_a=*/false, a, /*neg_b=*/false, b,
      /*c_mod=*/(short)0, c, /*reuse_a=*/false, /*reuse_b=*/false);
}

// ---------------------------------------------------------------------------
// Kernel 1: X16[b,n,c] = f16( x[b,n,c] + emb[t,c] ),  t = n / 256
// ---------------------------------------------------------------------------
__global__ __launch_bounds__(256) void k_add_emb_cvt(
    const float* __restrict__ x, const float* __restrict__ emb,
    _Float16* __restrict__ X16) {
  int i = blockIdx.x * 256 + threadIdx.x;           // [0, 6144*1024)
  int c  = i & (C_DIM - 1);
  int bn = i >> 10;
  int n  = bn % N_TOK;
  int t  = n >> 8;                                   // / (H*W)
  X16[i] = (_Float16)(x[i] + emb[t * C_DIM + c]);
}

// ---------------------------------------------------------------------------
// Kernel 2: generic f32 -> f16 convert
// ---------------------------------------------------------------------------
__global__ __launch_bounds__(256) void k_cvt_f16(
    const float* __restrict__ src, _Float16* __restrict__ dst, int total) {
  int i = blockIdx.x * 256 + threadIdx.x;
  if (i < total) dst[i] = (_Float16)src[i];
}

// ---------------------------------------------------------------------------
// Kernel 3: QKV GEMM.  qkv[m,j] = sum_k X16[m,k]*Wqkv[j,k] + b[j]
//   M = 6144, Nj = 3072, K = 1024.  32x32 macro-tile per wave (2x2 WMMA
//   tiles), 4 WMMAs per 32-wide k-step.  Epilogue scatters:
//   part0 -> Q*0.125 (b,h,n,d), part1 -> K (b,h,n,d), part2 -> V^T (b,h,d,n).
// ---------------------------------------------------------------------------
__global__ __launch_bounds__(128) void k_qkv_gemm(
    const _Float16* __restrict__ X, const _Float16* __restrict__ W,
    const float* __restrict__ bias,
    _Float16* __restrict__ Qo, _Float16* __restrict__ Ko,
    _Float16* __restrict__ Vt) {
  const int NTM = THREEC / 32;                       // 96 macro j-tiles
  int wave = threadIdx.x >> 5;
  int lane = threadIdx.x & 31;
  int tid  = blockIdx.x * 4 + wave;                  // 192*96 macro tiles
  int mt = tid / NTM, nt = tid % NTM;
  int m0 = mt * 32, j0 = nt * 32;
  int row = lane & 15;
  int hs  = lane >> 4;                               // lane-half select

  const _Float16* A0 = X + (size_t)(m0 + row) * C_DIM + hs * 8;
  const _Float16* A1 = A0 + (size_t)16 * C_DIM;
  const _Float16* B0 = W + (size_t)(j0 + row) * C_DIM + hs * 8;
  const _Float16* B1 = B0 + (size_t)16 * C_DIM;

  v8f acc[2][2] = {{v8f{}, v8f{}}, {v8f{}, v8f{}}};
#pragma unroll 2
  for (int k = 0; k < C_DIM; k += 32) {
    __builtin_prefetch(A0 + k + 256, 0, 3);          // global_prefetch_b8
    __builtin_prefetch(A1 + k + 256, 0, 3);
    __builtin_prefetch(B0 + k + 256, 0, 3);
    __builtin_prefetch(B1 + k + 256, 0, 3);
    v16h a0 = make_frag(A0 + k, A0 + k + 16);
    v16h a1 = make_frag(A1 + k, A1 + k + 16);
    v16h b0 = make_frag32(B0 + k);
    v16h b1 = make_frag32(B1 + k);
    acc[0][0] = wmma16(a0, b0, acc[0][0]);
    acc[0][1] = wmma16(a0, b1, acc[0][1]);
    acc[1][0] = wmma16(a1, b0, acc[1][0]);
    acc[1][1] = wmma16(a1, b1, acc[1][1]);
  }

#pragma unroll
  for (int ni = 0; ni < 2; ++ni) {
    int j = j0 + ni * 16 + row;
    float bv = bias[j];
    int part = j >> 10;                              // uniform per sub-tile
    int c = j & (C_DIM - 1);
    int h = c >> 6, d = c & (HD - 1);
#pragma unroll
    for (int mi = 0; mi < 2; ++mi) {
#pragma unroll
      for (int r = 0; r < 8; ++r) {
        int m = m0 + mi * 16 + r + hs * 8;
        int b_ = m / N_TOK;                          // uniform per sub-tile
        int n  = m - b_ * N_TOK;
        float val = acc[mi][ni][r] + bv;
        size_t bh = (size_t)(b_ * NH + h);
        if (part == 0) {
          Qo[(bh * N_TOK + n) * HD + d] = (_Float16)(val * 0.125f);
        } else if (part == 1) {
          Ko[(bh * N_TOK + n) * HD + d] = (_Float16)val;
        } else {
          Vt[(bh * HD + d) * N_TOK + n] = (_Float16)val;
        }
      }
    }
  }
}

// ---------------------------------------------------------------------------
// Kernel 4: flash attention. One wave per 16-query tile; stream 3072 keys in
// 32-key steps with online softmax; O accumulated 16x64 in f32.
// ---------------------------------------------------------------------------
__global__ __launch_bounds__(256) void k_attn(
    const _Float16* __restrict__ Q, const _Float16* __restrict__ Kt,
    const _Float16* __restrict__ Vt, _Float16* __restrict__ AO) {
  __shared__ __align__(16) _Float16 Pl[8][16][32];   // per-wave P bounce

  int wave = threadIdx.x >> 5;
  int lane = threadIdx.x & 31;
  int row  = lane & 15;
  int hs   = lane >> 4;

  int gq  = blockIdx.x * 8 + wave;                   // 6144 q-tiles
  int b_  = gq / (NH * 192);
  int rem = gq - b_ * (NH * 192);
  int h   = rem / 192;
  int qt  = rem - h * 192;

  size_t bh = (size_t)(b_ * NH + h);
  const _Float16* Qb = Q  + bh * N_TOK * HD;
  const _Float16* Kb = Kt + bh * N_TOK * HD;
  const _Float16* Vb = Vt + bh * HD * N_TOK;

  // Q fragments (reused for every key step): d = 0..31 and 32..63
  const _Float16* qp = Qb + (size_t)(qt * 16 + row) * HD + hs * 8;
  v16h aQ0 = make_frag(qp,      qp + 16);
  v16h aQ1 = make_frag(qp + 32, qp + 48);

  float m_r[8], l_r[8];
  v8f O[4] = {v8f{}, v8f{}, v8f{}, v8f{}};
#pragma unroll
  for (int r = 0; r < 8; ++r) { m_r[r] = -1e30f; l_r[r] = 0.0f; }

  for (int kb = 0; kb < N_TOK; kb += 32) {
    // S = Q * K^T for keys [kb, kb+32)
    const _Float16* k0 = Kb + (size_t)(kb + row) * HD + hs * 8;
    const _Float16* k1 = k0 + 16 * HD;
    v8f S0 = {}, S1 = {};
    S0 = wmma16(aQ0, make_frag32(k0),      S0);
    S0 = wmma16(aQ1, make_frag32(k0 + 32), S0);
    S1 = wmma16(aQ0, make_frag32(k1),      S1);
    S1 = wmma16(aQ1, make_frag32(k1 + 32), S1);

    // Online softmax update (row = r + hs*8, key dim across 16-lane half)
    float alpha[8];
#pragma unroll
    for (int r = 0; r < 8; ++r) {
      float mx = fmaxf(S0[r], S1[r]);
      mx = fmaxf(mx, __shfl_xor(mx, 1, 32));
      mx = fmaxf(mx, __shfl_xor(mx, 2, 32));
      mx = fmaxf(mx, __shfl_xor(mx, 4, 32));
      mx = fmaxf(mx, __shfl_xor(mx, 8, 32));
      float mn = fmaxf(m_r[r], mx);
      alpha[r] = __expf(m_r[r] - mn);
      m_r[r]   = mn;
      float p0 = __expf(S0[r] - mn);
      float p1 = __expf(S1[r] - mn);
      S0[r] = p0; S1[r] = p1;
      float s = p0 + p1;
      s += __shfl_xor(s, 1, 32);
      s += __shfl_xor(s, 2, 32);
      s += __shfl_xor(s, 4, 32);
      s += __shfl_xor(s, 8, 32);
      l_r[r] = l_r[r] * alpha[r] + s;
    }

    // Relayout P (D-layout -> A-layout) through LDS, wave-local.
#pragma unroll
    for (int r = 0; r < 8; ++r) {
      int mrow = r + hs * 8;
      Pl[wave][mrow][row]      = (_Float16)S0[r];
      Pl[wave][mrow][16 + row] = (_Float16)S1[r];
    }
    asm volatile("s_wait_dscnt 0" ::: "memory");
    const _Float16* pp = &Pl[wave][row][hs * 8];
    v16h aP = make_frag(pp, pp + 16);

    // Rescale O and accumulate P * V
#pragma unroll
    for (int dt = 0; dt < 4; ++dt) {
#pragma unroll
      for (int r = 0; r < 8; ++r) O[dt][r] *= alpha[r];
    }
#pragma unroll
    for (int dt = 0; dt < 4; ++dt) {
      const _Float16* vp = Vb + (size_t)(dt * 16 + row) * N_TOK + kb + hs * 8;
      O[dt] = wmma16(aP, make_frag32(vp), O[dt]);
    }
  }

  // Normalize and write per-head output back into (b, n, C) f16 layout.
#pragma unroll
  for (int dt = 0; dt < 4; ++dt) {
#pragma unroll
    for (int r = 0; r < 8; ++r) {
      float v = O[dt][r] / l_r[r];
      int qrow = qt * 16 + r + hs * 8;
      AO[((size_t)(b_ * N_TOK) + qrow) * C_DIM + h * HD + dt * 16 + row] =
          (_Float16)v;
    }
  }
}

// ---------------------------------------------------------------------------
// Kernel 5: output projection. out[m,j] = sum_k AO[m,k]*Wp[j,k] + pb[j]
//   M = 6144, Nj = 1024, K = 1024.  32x32 macro-tile per wave, f32 result
//   straight to d_out.
// ---------------------------------------------------------------------------
__global__ __launch_bounds__(128) void k_proj_gemm(
    const _Float16* __restrict__ A, const _Float16* __restrict__ W,
    const float* __restrict__ bias, float* __restrict__ out) {
  int wave = threadIdx.x >> 5;
  int lane = threadIdx.x & 31;
  int tid  = blockIdx.x * 4 + wave;                  // 192 * 32 macro tiles
  int mt = tid >> 5, nt = tid & 31;
  int m0 = mt * 32, j0 = nt * 32;
  int row = lane & 15;
  int hs  = lane >> 4;

  const _Float16* A0 = A + (size_t)(m0 + row) * C_DIM + hs * 8;
  const _Float16* A1 = A0 + (size_t)16 * C_DIM;
  const _Float16* B0 = W + (size_t)(j0 + row) * C_DIM + hs * 8;
  const _Float16* B1 = B0 + (size_t)16 * C_DIM;

  v8f acc[2][2] = {{v8f{}, v8f{}}, {v8f{}, v8f{}}};
#pragma unroll 2
  for (int k = 0; k < C_DIM; k += 32) {
    __builtin_prefetch(A0 + k + 256, 0, 3);
    __builtin_prefetch(A1 + k + 256, 0, 3);
    __builtin_prefetch(B0 + k + 256, 0, 3);
    __builtin_prefetch(B1 + k + 256, 0, 3);
    v16h a0 = make_frag(A0 + k, A0 + k + 16);
    v16h a1 = make_frag(A1 + k, A1 + k + 16);
    v16h b0 = make_frag32(B0 + k);
    v16h b1 = make_frag32(B1 + k);
    acc[0][0] = wmma16(a0, b0, acc[0][0]);
    acc[0][1] = wmma16(a0, b1, acc[0][1]);
    acc[1][0] = wmma16(a1, b0, acc[1][0]);
    acc[1][1] = wmma16(a1, b1, acc[1][1]);
  }

#pragma unroll
  for (int ni = 0; ni < 2; ++ni) {
    float bv = bias[j0 + ni * 16 + row];
#pragma unroll
    for (int mi = 0; mi < 2; ++mi) {
#pragma unroll
      for (int r = 0; r < 8; ++r) {
        int m = m0 + mi * 16 + r + hs * 8;
        out[(size_t)m * C_DIM + j0 + ni * 16 + row] = acc[mi][ni][r] + bv;
      }
    }
  }
}

// ---------------------------------------------------------------------------
// Host launcher. Workspace layout (bytes):
//   [0,        12.6M) X16  f16 (B*N, C)
//   [12.6M,    18.9M) Wqkv f16 (3C, C)
//   [18.9M,    21.0M) Wproj f16 (C, C)
//   [21.0M,    33.6M) Q    f16 (B,nh,N,hd)  (pre-scaled by 1/8)
//   [33.6M,    46.1M) K    f16 (B,nh,N,hd)
//   [46.1M,    58.7M) V^T  f16 (B,nh,hd,N)
//   [58.7M,    71.3M) AO   f16 (B*N, C)
// ---------------------------------------------------------------------------
extern "C" void kernel_launch(void* const* d_in, const int* in_sizes, int n_in,
                              void* d_out, int out_size, void* d_ws,
                              size_t ws_size, hipStream_t stream) {
  (void)in_sizes; (void)n_in; (void)out_size; (void)ws_size;
  const float* x      = (const float*)d_in[0];
  const float* emb    = (const float*)d_in[1];
  const float* qkv_w  = (const float*)d_in[2];
  const float* qkv_b  = (const float*)d_in[3];
  const float* proj_w = (const float*)d_in[4];
  const float* proj_b = (const float*)d_in[5];
  float* out = (float*)d_out;

  char* ws = (char*)d_ws;
  _Float16* X16   = (_Float16*)(ws);
  _Float16* Wqkv  = (_Float16*)(ws + 12582912);
  _Float16* Wproj = (_Float16*)(ws + 18874368);
  _Float16* Qb    = (_Float16*)(ws + 20971520);
  _Float16* Kb    = (_Float16*)(ws + 33554432);
  _Float16* Vt    = (_Float16*)(ws + 46137344);
  _Float16* AO    = (_Float16*)(ws + 58720256);

  // 1) x + temporal embedding, convert to f16 (6144*1024 elems)
  k_add_emb_cvt<<<24576, 256, 0, stream>>>(x, emb, X16);
  // 2) weight converts
  k_cvt_f16<<<12288, 256, 0, stream>>>(qkv_w, Wqkv, THREEC * C_DIM);
  k_cvt_f16<<<4096,  256, 0, stream>>>(proj_w, Wproj, C_DIM * C_DIM);
  // 3) QKV projection: 192*96 macro tiles, 4 waves/block
  k_qkv_gemm<<<4608, 128, 0, stream>>>(X16, Wqkv, qkv_b, Qb, Kb, Vt);
  // 4) flash attention: 6144 q-tiles, 8 waves/block
  k_attn<<<768, 256, 0, stream>>>(Qb, Kb, Vt, AO);
  // 5) output projection: 192*32 macro tiles, 4 waves/block
  k_proj_gemm<<<1536, 128, 0, stream>>>(AO, Wproj, proj_b, out);
}
